// TriangleMultiplicationOutgoing_40501541601301
// MI455X (gfx1250) — matrix operand order repaired and checked by reading, not compiled, not run on hardware
//
#include <hip/hip_runtime.h>
#include <hip/hip_bf16.h>
#include <math.h>

// ---------------------------------------------------------------------------
// TriangleMultiplicationOutgoing, MI455X (gfx1250, wave32, WMMA f16->f32)
// L = 512, C_z = C_h = 128
// ---------------------------------------------------------------------------

typedef _Float16 f16;
typedef __attribute__((ext_vector_type(16))) _Float16 v16h;
typedef __attribute__((ext_vector_type(8)))  _Float16 v8h;
typedef __attribute__((ext_vector_type(8)))  float    v8f;

#define LSZ   512
#define CSZ   128
#define PLANE ((size_t)LSZ * (size_t)LSZ)   // elements per channel plane

static __device__ __forceinline__ v16h cat8(v8h lo, v8h hi) {
  return __builtin_shufflevector(lo, hi, 0,1,2,3,4,5,6,7,8,9,10,11,12,13,14,15);
}

static __device__ __forceinline__ v8f wmma_f16f32(v16h a, v16h b, v8f c) {
  // D = A(16x32 f16) x B(32x16 f16) + C(16x16 f32)
  return __builtin_amdgcn_wmma_f32_16x16x32_f16(false, a, false, b,
                                                (short)0, c, false, false);
}

static __device__ __forceinline__ float sigmoidf(float x) {
  return 1.0f / (1.0f + __expf(-x));
}

// ---------------------------------------------------------------------------
// Kernel 0: convert the six 128x128 weight matrices to f16 (once, L2 resident)
// layout: [0]=Wap [1]=Wag [2]=Wbp [3]=Wbg [4]=Wg [5]=Wo
// ---------------------------------------------------------------------------
__global__ __launch_bounds__(256) void k_cvt_weights(
    const float* __restrict__ Wap, const float* __restrict__ Wag,
    const float* __restrict__ Wbp, const float* __restrict__ Wbg,
    const float* __restrict__ Wg,  const float* __restrict__ Wo,
    f16* __restrict__ dst) {
  int idx = blockIdx.x * 256 + threadIdx.x;      // 6*16384
  int m = idx >> 14, e = idx & 16383;
  const float* src = (m == 0) ? Wap : (m == 1) ? Wag : (m == 2) ? Wbp
                   : (m == 3) ? Wbg : (m == 4) ? Wg  : Wo;
  dst[idx] = (f16)src[e];
}

// ---------------------------------------------------------------------------
// Kernel 1: vk[i,j] = rsqrt(max(sum_k pm[i,k]*pm[j,k], 1))
// ---------------------------------------------------------------------------
__global__ __launch_bounds__(256) void k_validk(const float* __restrict__ pm,
                                                float* __restrict__ vk) {
  int idx = blockIdx.x * 256 + threadIdx.x;      // 512*512
  int i = idx >> 9, j = idx & 511;
  const float* pi = pm + (size_t)i * LSZ;
  const float* pj = pm + (size_t)j * LSZ;
  float s = 0.0f;
#pragma unroll 8
  for (int k = 0; k < LSZ; ++k) s += pi[k] * pj[k];
  vk[idx] = rsqrtf(fmaxf(s, 1.0f));
}

// ---------------------------------------------------------------------------
// Kernel 2: LayerNorm(z) + 5 fused projections + gating/mask, stores
// a, b, gate as f16 into channel-major planes ws[c][i*512+k].
// One block = 16 flattened (i,k) rows; 8 waves each own a 16-column slice.
// ---------------------------------------------------------------------------
__global__ __launch_bounds__(256) void k_proj(
    const float* __restrict__ z,   const float* __restrict__ pm,
    const float* __restrict__ lng, const float* __restrict__ lnb,
    const f16*  __restrict__ wf16,
    const float* __restrict__ bap, const float* __restrict__ bag,
    const float* __restrict__ bbp, const float* __restrict__ bbg,
    const float* __restrict__ bgg,
    f16* __restrict__ aWs, f16* __restrict__ bWs, f16* __restrict__ gWs) {
  __shared__ float zbuf[16 * CSZ];
  __shared__ f16   znbuf[16 * CSZ];
  __shared__ float maskbuf[16];

  const int t    = threadIdx.x;
  const int row0 = blockIdx.x * 16;              // flattened (i,k) base

  // ---- stage 16 rows of z (2048 f32) ----
  const float4* zsrc = (const float4*)(z + (size_t)row0 * CSZ);
  float4* zd = (float4*)zbuf;
  zd[t]       = zsrc[t];
  zd[t + 256] = zsrc[t + 256];
  if (t < 16) maskbuf[t] = pm[row0 + t];
  __syncthreads();

  // ---- LayerNorm: 16 lanes per row, shfl_xor tree within 16-lane groups ----
  {
    const int lane = t & 31;
    const int sub  = lane & 15;
    const int row  = ((t >> 5) << 1) + (lane >> 4);
    float vals[8], s = 0.0f, s2 = 0.0f;
#pragma unroll
    for (int m = 0; m < 8; ++m) {
      float v = zbuf[row * CSZ + sub + m * 16];
      vals[m] = v; s += v; s2 += v * v;
    }
#pragma unroll
    for (int d = 1; d < 16; d <<= 1) {
      s  += __shfl_xor(s,  d, 16);
      s2 += __shfl_xor(s2, d, 16);
    }
    const float mu  = s  * (1.0f / 128.0f);
    const float var = s2 * (1.0f / 128.0f) - mu * mu;
    const float rs  = rsqrtf(var + 1e-5f);
#pragma unroll
    for (int m = 0; m < 8; ++m) {
      int c = sub + m * 16;
      znbuf[row * CSZ + c] = (f16)((vals[m] - mu) * rs * lng[c] + lnb[c]);
    }
  }
  __syncthreads();

  // ---- 5 projections via WMMA: out[r, col] = sum_c zn[r,c] * W[col,c] ----
  const int wave = t >> 5, lane = t & 31;
  const int col  = (wave << 4) + (lane & 15);    // output column (0..127)
  const int arow = lane & 15;                    // A-operand row (M)
  const int asel = lane >> 4;                    // k-chunk selector
  const int rb   = asel << 3;                    // row base for C/D elements

  v8f acc[5];
#pragma unroll
  for (int p = 0; p < 5; ++p) acc[p] = {};

#pragma unroll
  for (int ks = 0; ks < 4; ++ks) {
    const int kb = ks * 32;
    // A operand: 16-bit 16x32 layout -> two contiguous 16B chunks per lane
    v8h lo = *(const v8h*)(znbuf + arow * CSZ + kb + asel * 16);
    v8h hi = *(const v8h*)(znbuf + arow * CSZ + kb + asel * 16 + 8);
    v16h A = cat8(lo, hi);
#pragma unroll
    for (int p = 0; p < 5; ++p) {
      // B operand: lane n holds 16 consecutive k for column `col` (32B)
      v16h B = *(const v16h*)(wf16 + (size_t)p * 16384 + col * CSZ
                              + kb + asel * 16);
      acc[p] = wmma_f16f32(A, B, acc[p]);
    }
  }

  // ---- epilogue: bias + sigmoid gates + pair mask, pack to f16 ----
  const float bapv = bap[col], bagv = bag[col];
  const float bbpv = bbp[col], bbgv = bbg[col], bggv = bgg[col];
  v8h av, bv, gv;
#pragma unroll
  for (int e = 0; e < 8; ++e) {
    float mk = maskbuf[e + rb];
    float a  = (acc[0][e] + bapv) * sigmoidf(acc[1][e] + bagv) * mk;
    float b  = (acc[2][e] + bbpv) * sigmoidf(acc[3][e] + bbgv) * mk;
    float g  = sigmoidf(acc[4][e] + bggv);
    av[e] = (f16)a; bv[e] = (f16)b; gv[e] = (f16)g;
  }
  // channel plane [c][i*512+k]; 8 consecutive k per thread -> one 16B store
  const size_t base = (size_t)col * PLANE + (size_t)row0 + rb;
  *(v8h*)(aWs + base) = av;
  *(v8h*)(bWs + base) = bv;
  *(v8h*)(gWs + base) = gv;
}

// ---------------------------------------------------------------------------
// Kernel 3: per-channel einsum x[i,j,c] = sum_k a[i,k,c] b[j,k,c] (WMMA),
// scale by vk, LayerNorm over c, out-projection (WMMA vs f16 Wo), gate, mask.
// One block = 16x16 (i,j) tile x all 128 channels. 512 threads = 16 waves.
// Dynamic LDS: 128KB only — xbuf f32[256][128]; LayerNorm converts each row
// to f16 IN PLACE (f16 row lives in the first 256B of its 512B f32 row), so
// two workgroups fit per WGP (256KB of 320KB) for latency hiding.
// ---------------------------------------------------------------------------
__global__ __launch_bounds__(512) void k_tri(
    const f16*  __restrict__ aWs, const f16* __restrict__ bWs,
    const f16*  __restrict__ gWs, const float* __restrict__ vk,
    const float* __restrict__ pm,
    const float* __restrict__ lng, const float* __restrict__ lnb,
    const f16*  __restrict__ wo16, const float* __restrict__ bo,
    float* __restrict__ out) {
  extern __shared__ char smem[];
  float* xbuf = (float*)smem;                      // [256][128] f32 (128KB)

  const int t    = threadIdx.x;
  const int wave = t >> 5, lane = t & 31;
  const int i0 = (blockIdx.x >> 5) << 4;
  const int j0 = (blockIdx.x & 31) << 4;
  const int arow = lane & 15, asel = lane >> 4, rb = asel << 3;
  const int jj0 = lane & 15;

  // vk scale per D element (same for all channels of this tile)
  float vks[8];
#pragma unroll
  for (int e = 0; e < 8; ++e)
    vks[e] = vk[(size_t)(i0 + e + rb) * LSZ + (j0 + jj0)];

  // ---- einsum: wave w owns channels 8w..8w+7, 16 K-steps each.
  //      Register double-buffer: K-step n+1 loads issued before WMMA n. ----
  for (int cc = 0; cc < 8; ++cc) {
    const int c = wave * 8 + cc;
    const f16* aprow = aWs + (size_t)c * PLANE + (size_t)(i0 + arow) * LSZ;
    const f16* bprow = bWs + (size_t)c * PLANE + (size_t)(j0 + arow) * LSZ;
    if (cc < 7) {                                  // hint next channel plane
      __builtin_prefetch(aprow + PLANE, 0, 1);
      __builtin_prefetch(bprow + PLANE, 0, 1);
    }
    v8h  lo = *(const v8h*)(aprow + asel * 16);
    v8h  hi = *(const v8h*)(aprow + asel * 16 + 8);
    v16h B  = *(const v16h*)(bprow + asel * 16);
    v8f acc = {};
#pragma unroll
    for (int ks = 0; ks < 16; ++ks) {
      v8h lo_n = {}, hi_n = {};
      v16h B_n = {};
      if (ks < 15) {
        const int kb = (ks + 1) * 32;
        lo_n = *(const v8h*)(aprow + kb + asel * 16);
        hi_n = *(const v8h*)(aprow + kb + asel * 16 + 8);
        B_n  = *(const v16h*)(bprow + kb + asel * 16);
      }
      acc = wmma_f16f32(cat8(lo, hi), B, acc);
      lo = lo_n; hi = hi_n; B = B_n;
    }
#pragma unroll
    for (int e = 0; e < 8; ++e) {
      int p = (e + rb) * 16 + jj0;                 // position within tile
      xbuf[p * CSZ + c] = acc[e] * vks[e];
    }
  }
  __syncthreads();

  // ---- LayerNorm across channels; convert to f16 in place.
  //      Chunked ascending read-then-write never clobbers unread f32 data:
  //      chunk k reads floats [8k,8k+8), writes floats [4k,4k+4). ----
  if (t < 256) {
    float* xr = xbuf + t * CSZ;
    float s = 0.0f, s2 = 0.0f;
#pragma unroll 8
    for (int c = 0; c < CSZ; ++c) { float v = xr[c]; s += v; s2 += v * v; }
    const float mu  = s  * (1.0f / 128.0f);
    const float var = s2 * (1.0f / 128.0f) - mu * mu;
    const float rs  = rsqrtf(var + 1e-5f);
    f16* xn = (f16*)xr;
    for (int c0 = 0; c0 < CSZ; c0 += 8) {
      float v[8];
#pragma unroll
      for (int e = 0; e < 8; ++e) v[e] = xr[c0 + e];
      asm volatile("" ::: "memory");               // order reads before writes
#pragma unroll
      for (int e = 0; e < 8; ++e)
        xn[c0 + e] = (f16)((v[e] - mu) * rs * lng[c0 + e] + lnb[c0 + e]);
    }
  }
  __syncthreads();

  // ---- out-projection: 16 row-tiles x 8 col-tiles; wave -> 8 tiles.
  //      xn rows are f16 at stride 256 elements (512B). ----
  const f16* xnb = (const f16*)xbuf;
  const int ct = wave & 7;
  const int rtBase = (wave >> 3) << 3;
  const int oc = ct * 16 + (lane & 15);
  const float bov = bo[oc];
  const f16* gplane = gWs + (size_t)oc * PLANE;

  v16h WoB[4];                                     // loop-invariant B tiles
#pragma unroll
  for (int ks = 0; ks < 4; ++ks)
    WoB[ks] = *(const v16h*)(wo16 + oc * CSZ + ks * 32 + asel * 16);

  for (int r = 0; r < 8; ++r) {
    const int rt = rtBase + r;                     // ii of this row-tile
    v8f acc = {};
#pragma unroll
    for (int ks = 0; ks < 4; ++ks) {
      const int kb = ks * 32;
      const f16* xrow = xnb + (rt * 16 + arow) * 256;
      v8h lo = *(const v8h*)(xrow + kb + asel * 16);
      v8h hi = *(const v8h*)(xrow + kb + asel * 16 + 8);
      acc = wmma_f16f32(cat8(lo, hi), WoB[ks], acc);
    }
    // gate: contiguous 8 x f16 along j in the gate channel plane
    v8h gvec = *(const v8h*)(gplane + (size_t)(i0 + rt) * LSZ + j0 + rb);
    const float* pmrow = pm + (size_t)(i0 + rt) * LSZ + j0 + rb;
    float* orow = out + ((size_t)(i0 + rt) * LSZ + j0 + rb) * CSZ + oc;
#pragma unroll
    for (int e = 0; e < 8; ++e)
      orow[(size_t)e * CSZ] = (acc[e] + bov) * (float)gvec[e] * pmrow[e];
  }
}

// ---------------------------------------------------------------------------
// launch
// ---------------------------------------------------------------------------
extern "C" void kernel_launch(void* const* d_in, const int* in_sizes, int n_in,
                              void* d_out, int out_size, void* d_ws, size_t ws_size,
                              hipStream_t stream) {
  const float* z    = (const float*)d_in[0];
  const float* pm   = (const float*)d_in[1];
  const float* lnig = (const float*)d_in[2];
  const float* lnib = (const float*)d_in[3];
  const float* Wap  = (const float*)d_in[4];   const float* bap = (const float*)d_in[5];
  const float* Wbp  = (const float*)d_in[6];   const float* bbp = (const float*)d_in[7];
  const float* Wag  = (const float*)d_in[8];   const float* bag = (const float*)d_in[9];
  const float* Wbg  = (const float*)d_in[10];  const float* bbg = (const float*)d_in[11];
  const float* lnog = (const float*)d_in[12];  const float* lnob = (const float*)d_in[13];
  const float* Wo   = (const float*)d_in[14];  const float* bo  = (const float*)d_in[15];
  const float* Wg   = (const float*)d_in[16];  const float* bg  = (const float*)d_in[17];
  float* outp = (float*)d_out;

  char* ws = (char*)d_ws;
  const size_t planeB = PLANE * sizeof(f16);           // 512KB per channel
  f16*   wf16 = (f16*)ws;                              // 6 * 32KB
  float* vkp  = (float*)(ws + (256ull << 10));         // 1MB  @ 256KB
  f16*   aWs  = (f16*)(ws + (2ull << 20));             // 64MB @ 2MB
  f16*   bWs  = (f16*)(ws + (2ull << 20) + 128 * planeB);
  f16*   gWs  = (f16*)(ws + (2ull << 20) + 256 * planeB);

  k_cvt_weights<<<dim3(6 * 16384 / 256), dim3(256), 0, stream>>>(
      Wap, Wag, Wbp, Wbg, Wg, Wo, wf16);

  k_validk<<<dim3(LSZ * LSZ / 256), dim3(256), 0, stream>>>(pm, vkp);

  k_proj<<<dim3(LSZ * LSZ / 16), dim3(256), 0, stream>>>(
      z, pm, lnig, lnib, wf16, bap, bag, bbp, bbg, bg, aWs, bWs, gWs);

  const size_t smem = 256 * CSZ * 4;                   // 128KB dynamic LDS
  k_tri<<<dim3((LSZ / 16) * (LSZ / 16)), dim3(512), smem, stream>>>(
      aWs, bWs, gWs, vkp, pm, lnog, lnob, wf16 + 5 * 16384, bo, outp);
}
